// GATSelector_83159156785733
// MI455X (gfx1250) — matrix operand order repaired and checked by reading, not compiled, not run on hardware
//
#include <hip/hip_runtime.h>
#include <hip/hip_bf16.h>
#include <math.h>

typedef __attribute__((ext_vector_type(16))) _Float16 v16h;
typedef __attribute__((ext_vector_type(8)))  _Float16 v8h;
typedef __attribute__((ext_vector_type(8)))  float    v8f;

#define NEG_SLOPE 0.2f
#define FIN 128   // layer-1 input features == H*C1 == 128

// ---------------- utility: zero fill ----------------
__global__ void k_zero(float* __restrict__ p, long long n) {
  long long i = (long long)blockIdx.x * blockDim.x + threadIdx.x;
  if (i < n) p[i] = 0.0f;
}

// ---------------- f32 -> f16 conversions ----------------
__global__ void k_cvt_x(const float* __restrict__ x, _Float16* __restrict__ xh, long long n) {
  long long i = (long long)blockIdx.x * blockDim.x + threadIdx.x;
  if (i < n) xh[i] = (_Float16)x[i];
}

// W1 is [128 (k) x 128 (col)] row-major; store transposed f16: w1t[col*128 + k]
__global__ void k_cvt_w1t(const float* __restrict__ W1, _Float16* __restrict__ w1t) {
  int i = blockIdx.x * blockDim.x + threadIdx.x;   // i = col*128 + k
  if (i >= FIN * FIN) return;
  int col = i >> 7, k = i & 127;
  w1t[i] = (_Float16)W1[k * FIN + col];
}

// ---------------- WMMA GEMM: h1 = x @ W1  (M=N nodes, N=128, K=128) ----------------
// grid.x = N/16 row tiles; block = 256 threads = 8 waves; wave w -> column tile w.
__global__ void k_gemm_wmma(const _Float16* __restrict__ xh,
                            const _Float16* __restrict__ w1t,
                            float* __restrict__ h1) {
  const int wave = threadIdx.x >> 5;      // 0..7 -> column tile
  const int lane = threadIdx.x & 31;
  const int rowTile = blockIdx.x;
  const int row = rowTile * 16 + (lane & 15);     // A: lane holds one row
  const int col = wave * 16 + (lane & 15);        // B: lane holds one column
  const int kbaseA = (lane >> 4) * 8;             // A 16-bit layout: K = kbase+{0..7}, kbase+16+{0..7}
  const int kbaseB = (lane >> 4) * 8;             // B 16-bit layout: K = kbase+{0..7}, kbase+16+{0..7}

  v8f acc = {};
#pragma unroll
  for (int k0 = 0; k0 < FIN; k0 += 32) {
    const v8h a_lo = *(const v8h*)(xh  + (size_t)row * FIN + k0 + kbaseA);
    const v8h a_hi = *(const v8h*)(xh  + (size_t)row * FIN + k0 + kbaseA + 16);
    const v8h b_lo = *(const v8h*)(w1t + (size_t)col * FIN + k0 + kbaseB);
    const v8h b_hi = *(const v8h*)(w1t + (size_t)col * FIN + k0 + kbaseB + 16);
    v16h A, B;
#pragma unroll
    for (int j = 0; j < 8; ++j) {
      A[j] = a_lo[j]; A[j + 8] = a_hi[j];
      B[j] = b_lo[j]; B[j + 8] = b_hi[j];
    }
    acc = __builtin_amdgcn_wmma_f32_16x16x32_f16(false, A, false, B,
                                                 (short)0, acc, false, false);
  }
  // D layout: col = lane&15, row M = e + 8*(lane>>4)
  const int nOut = wave * 16 + (lane & 15);
#pragma unroll
  for (int e = 0; e < 8; ++e) {
    int m = rowTile * 16 + e + 8 * (lane >> 4);
    h1[(size_t)m * FIN + nOut] = acc[e];
  }
}

// ---------------- attention coefficients per (node, head) ----------------
template <int H, int C>
__global__ void k_attn(const float* __restrict__ h1,
                       const float* __restrict__ att_src,
                       const float* __restrict__ att_dst,
                       float* __restrict__ a_s, float* __restrict__ a_d, int N) {
  int i = blockIdx.x * blockDim.x + threadIdx.x;    // i = n*H + h
  if (i >= N * H) return;
  int h = i % H;
  const float* hp = h1 + (size_t)i * C;             // [N,H,C] flat
  float ss = 0.f, sd = 0.f;
#pragma unroll
  for (int c = 0; c < C; ++c) {
    float v = hp[c];
    ss += v * att_src[h * C + c];
    sd += v * att_dst[h * C + c];
  }
  a_s[i] = ss; a_d[i] = sd;
}

// ---------------- float ordered-uint helpers for atomic max ----------------
__device__ __forceinline__ unsigned ford(float f) {
  unsigned u = __float_as_uint(f);
  return (u & 0x80000000u) ? ~u : (u | 0x80000000u);
}
__device__ __forceinline__ float finv(unsigned e) {
  unsigned u = (e & 0x80000000u) ? (e & 0x7FFFFFFFu) : ~e;
  return __uint_as_float(u);
}

__device__ __forceinline__ void edge_ends(const long long* __restrict__ ei,
                                          long long E, long long e, int* s, int* d) {
  if (e < E) { *s = (int)ei[e]; *d = (int)ei[E + e]; }
  else       { int n = (int)(e - E); *s = n; *d = n; }   // appended self loops
}

template <int H>
__device__ __forceinline__ float edge_logit(const float* __restrict__ a_s,
                                            const float* __restrict__ a_d,
                                            int s, int d, int h) {
  float ev = a_s[(size_t)s * H + h] + a_d[(size_t)d * H + h];
  return ev > 0.f ? ev : NEG_SLOPE * ev;
}

// ---------------- edge pass 1: segment max ----------------
template <int H>
__global__ void k_edge_max(const long long* __restrict__ ei, long long E, long long Etot,
                           const float* __restrict__ a_s, const float* __restrict__ a_d,
                           unsigned* __restrict__ emax) {
  long long i = (long long)blockIdx.x * blockDim.x + threadIdx.x;
  if (i >= Etot * H) return;
  long long e = i / H; int h = (int)(i - e * H);
  int s, d; edge_ends(ei, E, e, &s, &d);
  float ev = edge_logit<H>(a_s, a_d, s, d, h);
  atomicMax(&emax[(size_t)d * H + h], ford(ev));
}

// ---------------- edge pass 2: softmax denominator ----------------
template <int H>
__global__ void k_edge_den(const long long* __restrict__ ei, long long E, long long Etot,
                           const float* __restrict__ a_s, const float* __restrict__ a_d,
                           const unsigned* __restrict__ emax, float* __restrict__ den) {
  long long i = (long long)blockIdx.x * blockDim.x + threadIdx.x;
  if (i >= Etot * H) return;
  long long e = i / H; int h = (int)(i - e * H);
  int s, d; edge_ends(ei, E, e, &s, &d);
  float ev = edge_logit<H>(a_s, a_d, s, d, h);
  float ex = __expf(ev - finv(emax[(size_t)d * H + h]));
  atomicAdd(&den[(size_t)d * H + h], ex);
}

// ---------------- edge pass 3: weighted message aggregation ----------------
template <int H, int C>
__global__ void k_edge_msg(const long long* __restrict__ ei, long long E, long long Etot,
                           const float* __restrict__ a_s, const float* __restrict__ a_d,
                           const unsigned* __restrict__ emax, const float* __restrict__ den,
                           const float* __restrict__ feat,   // [N,H,C] source features
                           float* __restrict__ outacc) {     // [N,H,C]
  long long i = (long long)blockIdx.x * blockDim.x + threadIdx.x;
  if (i >= Etot * H) return;
  long long e = i / H; int h = (int)(i - e * H);
  int s, d; edge_ends(ei, E, e, &s, &d);
  float ev = edge_logit<H>(a_s, a_d, s, d, h);
  float ex = __expf(ev - finv(emax[(size_t)d * H + h]));
  float alpha = ex / (den[(size_t)d * H + h] + 1e-16f);
  const float* fs = feat + ((size_t)s * H + h) * C;
  float* od = outacc + ((size_t)d * H + h) * C;
  if constexpr ((C & 3) == 0) {
    // vectorized gather (rows are 16B-aligned: C multiple of 4, base 256B-aligned)
    const float4* fs4 = (const float4*)fs;
#pragma unroll
    for (int g = 0; g < C / 4; ++g) {
      float4 v = fs4[g];
      atomicAdd(&od[g * 4 + 0], v.x * alpha);
      atomicAdd(&od[g * 4 + 1], v.y * alpha);
      atomicAdd(&od[g * 4 + 2], v.z * alpha);
      atomicAdd(&od[g * 4 + 3], v.w * alpha);
    }
  } else {
#pragma unroll
    for (int c = 0; c < C; ++c) atomicAdd(&od[c], fs[c] * alpha);
  }
}

// ---------------- bias + relu (layer 1 epilogue) ----------------
__global__ void k_bias_relu(float* __restrict__ o, const float* __restrict__ b, long long n) {
  long long i = (long long)blockIdx.x * blockDim.x + threadIdx.x;
  if (i >= n) return;
  float v = o[i] + b[(int)(i & (FIN - 1))];
  o[i] = v > 0.f ? v : 0.f;
}

// ---------------- layer 2 node transform: h2 = out1 @ W2 (128 -> 1) ----------------
__global__ void k_l2_node(const float* __restrict__ out1, const float* __restrict__ W2,
                          const float* __restrict__ att_s2, const float* __restrict__ att_d2,
                          float* __restrict__ h2, float* __restrict__ a_s2,
                          float* __restrict__ a_d2, int N) {
  int n = blockIdx.x * blockDim.x + threadIdx.x;
  if (n >= N) return;
  const float4* r = (const float4*)(out1 + (size_t)n * FIN);
  const float4* w = (const float4*)W2;
  float s = 0.f;
#pragma unroll
  for (int k = 0; k < FIN / 4; ++k) {
    float4 a = r[k], bq = w[k];
    s += a.x * bq.x + a.y * bq.y + a.z * bq.z + a.w * bq.w;
  }
  h2[n] = s;
  a_s2[n] = s * att_s2[0];
  a_d2[n] = s * att_d2[0];
}

// ---------------- final: out = out2 + b2 ----------------
__global__ void k_final(const float* __restrict__ acc, const float* __restrict__ b2,
                        float* __restrict__ out, int N) {
  int n = blockIdx.x * blockDim.x + threadIdx.x;
  if (n < N) out[n] = acc[n] + b2[0];
}

static inline size_t align_up(size_t v, size_t a) { return (v + a - 1) & ~(a - 1); }

extern "C" void kernel_launch(void* const* d_in, const int* in_sizes, int n_in,
                              void* d_out, int out_size, void* d_ws, size_t ws_size,
                              hipStream_t stream) {
  (void)n_in; (void)out_size; (void)ws_size;
  const float*     x        = (const float*)d_in[0];
  const long long* ei       = (const long long*)d_in[1];
  const float*     W1       = (const float*)d_in[2];
  const float*     att_src1 = (const float*)d_in[3];
  const float*     att_dst1 = (const float*)d_in[4];
  const float*     b1       = (const float*)d_in[5];
  const float*     W2       = (const float*)d_in[6];
  const float*     att_src2 = (const float*)d_in[7];
  const float*     att_dst2 = (const float*)d_in[8];
  const float*     b2       = (const float*)d_in[9];

  const int  N    = in_sizes[0] / FIN;
  const long long E    = (long long)in_sizes[1] / 2;
  const long long Etot = E + N;
  constexpr int H1 = 4, C1 = 32;

  // ---- workspace carve ----
  char* ws = (char*)d_ws;
  size_t off = 0;
  auto carve = [&](size_t bytes) { size_t o = off; off = align_up(off + bytes, 256); return (void*)(ws + o); };
  _Float16* xh    = (_Float16*)carve((size_t)N * FIN * 2);
  _Float16* w1t   = (_Float16*)carve((size_t)FIN * FIN * 2);
  float*    h1    = (float*)carve((size_t)N * FIN * 4);
  float*    a_s1  = (float*)carve((size_t)N * H1 * 4);
  float*    a_d1  = (float*)carve((size_t)N * H1 * 4);
  unsigned* emax1 = (unsigned*)carve((size_t)N * H1 * 4);
  float*    den1  = (float*)carve((size_t)N * H1 * 4);
  float*    out1  = (float*)carve((size_t)N * FIN * 4);
  float*    h2    = (float*)carve((size_t)N * 4);
  float*    a_s2  = (float*)carve((size_t)N * 4);
  float*    a_d2  = (float*)carve((size_t)N * 4);
  unsigned* emax2 = (unsigned*)carve((size_t)N * 4);
  float*    den2  = (float*)carve((size_t)N * 4);
  float*    out2  = (float*)carve((size_t)N * 4);

  const int B = 256;
  auto blocks = [&](long long n) { return (unsigned)((n + B - 1) / B); };

  // zero-init accumulators (ordered-uint 0 == below every encoded float)
  k_zero<<<blocks((long long)N * H1), B, 0, stream>>>((float*)emax1, (long long)N * H1);
  k_zero<<<blocks((long long)N * H1), B, 0, stream>>>(den1, (long long)N * H1);
  k_zero<<<blocks((long long)N * FIN), B, 0, stream>>>(out1, (long long)N * FIN);
  k_zero<<<blocks(N), B, 0, stream>>>((float*)emax2, N);
  k_zero<<<blocks(N), B, 0, stream>>>(den2, N);
  k_zero<<<blocks(N), B, 0, stream>>>(out2, N);

  // ---- layer 1 ----
  k_cvt_x<<<blocks((long long)N * FIN), B, 0, stream>>>(x, xh, (long long)N * FIN);
  k_cvt_w1t<<<blocks(FIN * FIN), B, 0, stream>>>(W1, w1t);
  k_gemm_wmma<<<N / 16, 256, 0, stream>>>(xh, w1t, h1);
  k_attn<H1, C1><<<blocks((long long)N * H1), B, 0, stream>>>(h1, att_src1, att_dst1,
                                                              a_s1, a_d1, N);
  k_edge_max<H1><<<blocks(Etot * H1), B, 0, stream>>>(ei, E, Etot, a_s1, a_d1, emax1);
  k_edge_den<H1><<<blocks(Etot * H1), B, 0, stream>>>(ei, E, Etot, a_s1, a_d1, emax1, den1);
  k_edge_msg<H1, C1><<<blocks(Etot * H1), B, 0, stream>>>(ei, E, Etot, a_s1, a_d1,
                                                          emax1, den1, h1, out1);
  k_bias_relu<<<blocks((long long)N * FIN), B, 0, stream>>>(out1, b1, (long long)N * FIN);

  // ---- layer 2 (H=1, C=1) ----
  k_l2_node<<<blocks(N), B, 0, stream>>>(out1, W2, att_src2, att_dst2, h2, a_s2, a_d2, N);
  k_edge_max<1><<<blocks(Etot), B, 0, stream>>>(ei, E, Etot, a_s2, a_d2, emax2);
  k_edge_den<1><<<blocks(Etot), B, 0, stream>>>(ei, E, Etot, a_s2, a_d2, emax2, den2);
  k_edge_msg<1, 1><<<blocks(Etot), B, 0, stream>>>(ei, E, Etot, a_s2, a_d2, emax2, den2,
                                                   h2, out2);
  k_final<<<blocks(N), B, 0, stream>>>(out2, b2, (float*)d_out, N);
}
